// MultiHeadAttention_54537494724825
// MI455X (gfx1250) — compile-verified, matrix operand order checked by reading, not run on hardware
//
#include <hip/hip_runtime.h>
#include <hip/hip_bf16.h>

// ---------------------------------------------------------------------------
// MultiHeadAttention scores+softmax for MI455X (gfx1250, wave32, WMMA)
//
// Q = query @ Wq^T + bq ; K = key @ Wk^T + bk  (all-f16 GEMMs, f32 accum)
// energy[h,n,m] = Q[n,h,:] . K[m,h,:] ; out = softmax(energy, axis=-1)
// V projection intentionally skipped (unused in reference).
//
// Roofline: output write 604 MB (~26 us at 23.3 TB/s) dominates; Q/K/W f16
// copies (~67 MB) are L2-resident; ~90 GFLOP of f16 WMMA is comparable to
// the write floor, so everything else is organized to never touch HBM twice.
// ---------------------------------------------------------------------------

#define N_TOK   3072
#define D_MODEL 2048
#define N_HEADS 16
#define HEAD_DIM 128

typedef __attribute__((ext_vector_type(16))) _Float16 v16h;
typedef __attribute__((ext_vector_type(8)))  _Float16 v8h;
typedef __attribute__((ext_vector_type(8)))  float    v8f;
typedef int v4i __attribute__((vector_size(16)));     // matches async builtin param

#if __has_builtin(__builtin_amdgcn_global_load_async_to_lds_b128) && \
    __has_builtin(__builtin_amdgcn_s_wait_asynccnt)
#define USE_ASYNC_LDS 1
// clang prints AS1 as "__device__": param types are v4i in AS1 (global) / AS3 (LDS)
#define ASYNC_GPTR(p) ((__attribute__((address_space(1))) v4i*)(p))
#define ASYNC_LPTR(p) ((__attribute__((address_space(3))) v4i*)(p))
#else
#define USE_ASYNC_LDS 0
#endif

// ---- fragment loader ------------------------------------------------------
// 16-bit A/B fragment layout (ISA 7.12.2, 16x32 f16):
//   lanes 0-15  (row M = lane)   : K = {0..7} and {16..23}
//   lanes 16-31 (row M = lane-16): K = {8..15} and {24..31}
// => per lane: two contiguous 8-half (16B, b128) loads at k = base_k and
//    base_k+16, base_k = (lane>=16) ? 8 : 0.
__device__ __forceinline__ v16h frag_from_f16(const _Float16* __restrict__ p, int base_k) {
    v8h lo = *(const v8h*)(p + base_k);
    v8h hi = *(const v8h*)(p + base_k + 16);
    v16h r;
#pragma unroll
    for (int i = 0; i < 8; ++i) { r[i] = lo[i]; r[i + 8] = hi[i]; }
    return r;
}

// ---------------------------------------------------------------------------
// Streaming f32 -> f16 convert (keeps ALL cvt out of the GEMM hot loops).
// 8 elements/thread: one global_load_b128 x2 -> cvt -> global_store_b128.
// ---------------------------------------------------------------------------
__global__ __launch_bounds__(256)
void cvt_f32_to_f16_kernel(const float* __restrict__ src, _Float16* __restrict__ dst) {
    const size_t i = ((size_t)blockIdx.x * 256 + threadIdx.x) * 8;
    v8f v = *(const v8f*)(src + i);
    *(v8h*)(dst + i) = __builtin_convertvector(v, v8h);
}

// ---------------------------------------------------------------------------
// Projection GEMM (pure f16):  Y[n][d] = sum_k X[n][k] * W[d][k] + b[d].
// Block: 256 threads = 8 waves. Tile: 128 rows (16/wave) x 64 cols.
// W k-slab (64x32 f16 = 4 KB) double-buffered in LDS, filled by exactly one
// GLOBAL_LOAD_ASYNC_TO_LDS_B128 per thread per iter (ASYNCcnt DMA path);
// all fragment registers are written by memory ops -> no WMMA/VALU hazard NOPs.
// ---------------------------------------------------------------------------
__global__ __launch_bounds__(256)
void proj_kernel(const _Float16* __restrict__ X, const _Float16* __restrict__ W,
                 const float* __restrict__ B, _Float16* __restrict__ Y) {
    __shared__ _Float16 lds_w[2][64 * 32];        // 2 x 4 KB: W[d0..+63][k..k+31]

    const int tid  = threadIdx.x;
    const int wave = tid >> 5;
    const int lane = tid & 31;
    const int r    = lane & 15;
    const int half = lane >> 4;
    const int bk   = half << 3;                   // base_k: 0 or 8

    const int m0 = blockIdx.x * 128 + wave * 16;  // row tile of this wave
    const int d0 = blockIdx.y * 64;               // col tile of this block

    v8f acc[4];
#pragma unroll
    for (int nt = 0; nt < 4; ++nt) { v8f z = {}; acc[nt] = z; }

    const _Float16* xrow = X + (size_t)(m0 + r) * D_MODEL;
    const int fill_row = tid >> 2;                // 0..63
    const int fill_seg = (tid & 3) << 3;          // 0,8,16,24 (halves; 16B chunks)
    const _Float16* wfill = W + (size_t)(d0 + fill_row) * D_MODEL + fill_seg;

    // slab fill: 64x32 f16 = 4 KB; 256 threads x one async b128 each
    auto fill = [&](int buf, int kc) {
        const _Float16* g = wfill + kc * 32;
        _Float16*       l = &lds_w[buf][fill_row * 32 + fill_seg];
#if USE_ASYNC_LDS
        __builtin_amdgcn_global_load_async_to_lds_b128(ASYNC_GPTR(g), ASYNC_LPTR(l), 0, 0);
#else
        *(v8h*)l = *(const v8h*)g;
#endif
    };

    fill(0, 0);
    for (int kc = 0; kc < D_MODEL / 32; ++kc) {
        const int p = kc & 1;
#if USE_ASYNC_LDS
        __builtin_amdgcn_s_wait_asynccnt(0);      // this wave's DMA landed in LDS
#endif
        __syncthreads();                          // all fills done, prev reads done
        if (kc + 1 < D_MODEL / 32) fill(p ^ 1, kc + 1);

        v16h a = frag_from_f16(xrow + kc * 32, bk);
        v16h b[4];
#pragma unroll
        for (int nt = 0; nt < 4; ++nt)
            b[nt] = frag_from_f16(&lds_w[p][(nt * 16 + r) * 32], bk);
#pragma unroll
        for (int nt = 0; nt < 4; ++nt)
            acc[nt] = __builtin_amdgcn_wmma_f32_16x16x32_f16(
                false, a, false, b[nt], (short)0, acc[nt], false, false);
    }

    // epilogue: C layout (ISA 7.12.2): VGPR j -> row M = j + 8*half, col N = r
#pragma unroll
    for (int nt = 0; nt < 4; ++nt) {
        const int col  = d0 + nt * 16 + r;
        const float bv = B[col];
#pragma unroll
        for (int j = 0; j < 8; ++j) {
            const int row = m0 + j + (half << 3);
            Y[(size_t)row * D_MODEL + col] = (_Float16)(acc[nt][j] + bv);
        }
    }
}

// ---------------------------------------------------------------------------
// Fused scores + softmax (two-pass flash):
// Block: one head x 16 query rows, 8 waves sweep the 192 key tiles.
// Pass 1: WMMA scores with two independent accumulators; each lane keeps a
//         PRIVATE running (max, expsum) for its 8 rows -> no shuffles in the
//         hot loop. One flash-merge over 16 lanes (shfl_xor 1/2/4/8; masks
//         < 16 keep the two half-waves' row groups independent), then an
//         8-wave combine through LDS.
// Pass 2: recompute scores, stream exp(s - M) / S straight to d_out
//         (604 MB written exactly once -> pinned at the 23.3 TB/s write floor).
// ---------------------------------------------------------------------------
__global__ __launch_bounds__(256)
void scores_softmax_kernel(const _Float16* __restrict__ Qh,
                           const _Float16* __restrict__ Kh,
                           float* __restrict__ out) {
    __shared__ float smax[8][16];
    __shared__ float ssum[8][16];

    const int tid  = threadIdx.x;
    const int wave = tid >> 5;
    const int lane = tid & 31;
    const int r    = lane & 15;
    const int half = lane >> 4;
    const int bk   = half << 3;

    const int nblocks = N_TOK / 16;               // 192
    const int h  = blockIdx.x / nblocks;
    const int n0 = (blockIdx.x % nblocks) * 16;

    // Q fragments for this 16-row tile (A operand), resident for both passes
    const _Float16* qrow = Qh + (size_t)(n0 + r) * D_MODEL + h * HEAD_DIM;
    v16h qa[4];
#pragma unroll
    for (int kc = 0; kc < 4; ++kc) qa[kc] = frag_from_f16(qrow + kc * 32, bk);

    float lmax[8], lsum[8];
#pragma unroll
    for (int j = 0; j < 8; ++j) { lmax[j] = -3.402823466e38f; lsum[j] = 0.0f; }

    // ---------------- pass 1: per-lane online max / expsum ----------------
    for (int t = wave; t < nblocks; t += 8) {
        const _Float16* krow = Kh + (size_t)(t * 16 + r) * D_MODEL + h * HEAD_DIM;
        v16h kb[4];
#pragma unroll
        for (int kc = 0; kc < 4; ++kc) kb[kc] = frag_from_f16(krow + kc * 32, bk);
        v8f s0 = {}, s1 = {};
        s0 = __builtin_amdgcn_wmma_f32_16x16x32_f16(false, qa[0], false, kb[0], (short)0, s0, false, false);
        s1 = __builtin_amdgcn_wmma_f32_16x16x32_f16(false, qa[1], false, kb[1], (short)0, s1, false, false);
        s0 = __builtin_amdgcn_wmma_f32_16x16x32_f16(false, qa[2], false, kb[2], (short)0, s0, false, false);
        s1 = __builtin_amdgcn_wmma_f32_16x16x32_f16(false, qa[3], false, kb[3], (short)0, s1, false, false);
#pragma unroll
        for (int j = 0; j < 8; ++j) {
            const float v    = s0[j] + s1[j];
            const float mnew = fmaxf(lmax[j], v);
            lsum[j] = lsum[j] * __expf(lmax[j] - mnew) + __expf(v - mnew);
            lmax[j] = mnew;
        }
    }

    // flash-merge across the 16 lanes of each half-wave
#pragma unroll
    for (int j = 0; j < 8; ++j) {
        float m = lmax[j], s = lsum[j];
#pragma unroll
        for (int off = 1; off < 16; off <<= 1) {
            const float om = __shfl_xor(m, off, 32);
            const float os = __shfl_xor(s, off, 32);
            const float M  = fmaxf(m, om);
            s = s * __expf(m - M) + os * __expf(om - M);
            m = M;
        }
        lmax[j] = m; lsum[j] = s;
    }

    // ---------------- cross-wave combine ----------------
    if (r == 0) {
#pragma unroll
        for (int j = 0; j < 8; ++j) {
            smax[wave][j + (half << 3)] = lmax[j];
            ssum[wave][j + (half << 3)] = lsum[j];
        }
    }
    __syncthreads();

    float rowM[8], rowInv[8];
#pragma unroll
    for (int j = 0; j < 8; ++j) {
        const int row = j + (half << 3);
        float M = -3.402823466e38f;
#pragma unroll
        for (int w = 0; w < 8; ++w) M = fmaxf(M, smax[w][row]);
        float S = 0.0f;
#pragma unroll
        for (int w = 0; w < 8; ++w) S += ssum[w][row] * __expf(smax[w][row] - M);
        rowM[j]   = M;
        rowInv[j] = 1.0f / S;
    }

    // ---------------- pass 2: recompute + normalized store ----------------
    for (int t = wave; t < nblocks; t += 8) {
        const _Float16* krow = Kh + (size_t)(t * 16 + r) * D_MODEL + h * HEAD_DIM;
        v16h kb[4];
#pragma unroll
        for (int kc = 0; kc < 4; ++kc) kb[kc] = frag_from_f16(krow + kc * 32, bk);
        v8f s0 = {}, s1 = {};
        s0 = __builtin_amdgcn_wmma_f32_16x16x32_f16(false, qa[0], false, kb[0], (short)0, s0, false, false);
        s1 = __builtin_amdgcn_wmma_f32_16x16x32_f16(false, qa[1], false, kb[1], (short)0, s1, false, false);
        s0 = __builtin_amdgcn_wmma_f32_16x16x32_f16(false, qa[2], false, kb[2], (short)0, s0, false, false);
        s1 = __builtin_amdgcn_wmma_f32_16x16x32_f16(false, qa[3], false, kb[3], (short)0, s1, false, false);
        const int m0 = t * 16;
#pragma unroll
        for (int j = 0; j < 8; ++j) {
            const int row = n0 + j + (half << 3);
            out[((size_t)h * N_TOK + row) * N_TOK + m0 + r] =
                __expf((s0[j] + s1[j]) - rowM[j]) * rowInv[j];
        }
    }
}

// ---------------------------------------------------------------------------
extern "C" void kernel_launch(void* const* d_in, const int* in_sizes, int n_in,
                              void* d_out, int out_size, void* d_ws, size_t ws_size,
                              hipStream_t stream) {
    (void)in_sizes; (void)n_in; (void)out_size; (void)ws_size;
    const float* query = (const float*)d_in[0];
    const float* key   = (const float*)d_in[1];
    // d_in[2] (value), d_in[7] (Wv), d_in[8] (bv) unused: V is dead in reference
    const float* Wq = (const float*)d_in[3];
    const float* bq = (const float*)d_in[4];
    const float* Wk = (const float*)d_in[5];
    const float* bk = (const float*)d_in[6];

    const size_t XN = (size_t)N_TOK   * D_MODEL;          // 6.29 M elems
    const size_t WN = (size_t)D_MODEL * D_MODEL;          // 4.19 M elems

    _Float16* Qh  = (_Float16*)d_ws;                      // [N_TOK x D_MODEL] f16
    _Float16* Kh  = Qh  + XN;
    _Float16* Xqh = Kh  + XN;                             // f16 copy of query
    _Float16* Xkh = Xqh + XN;                             // f16 copy of key
    _Float16* Wqh = Xkh + XN;                             // f16 copy of Wq
    _Float16* Wkh = Wqh + WN;                             // f16 copy of Wk
    // total ws use: 4*12.6 MB + 2*8.4 MB = ~67 MB

    const int CV = 256 * 8;                               // elems per block
    cvt_f32_to_f16_kernel<<<XN / CV, 256, 0, stream>>>(query, Xqh);
    cvt_f32_to_f16_kernel<<<XN / CV, 256, 0, stream>>>(key,   Xkh);
    cvt_f32_to_f16_kernel<<<WN / CV, 256, 0, stream>>>(Wq,    Wqh);
    cvt_f32_to_f16_kernel<<<WN / CV, 256, 0, stream>>>(Wk,    Wkh);

    dim3 pgrid(N_TOK / 128, D_MODEL / 64);                // 24 x 32
    proj_kernel<<<pgrid, 256, 0, stream>>>(Xqh, Wqh, bq, Qh);
    proj_kernel<<<pgrid, 256, 0, stream>>>(Xkh, Wkh, bk, Kh);

    scores_softmax_kernel<<<N_HEADS * (N_TOK / 16), 256, 0, stream>>>(
        Qh, Kh, (float*)d_out);
}
